// VisionActionTransformer_22144851378577
// MI455X (gfx1250) — matrix operand order repaired and checked by reading, not compiled, not run on hardware
//
#include <hip/hip_runtime.h>
#include <hip/hip_bf16.h>

#define C_DIM 1152
#define HEADS 16
#define HD 72
#define TOK_ACT 57
#define BATCH 32
#define NTOK 825
#define NNON 768
#define GSZ 256
#define SCALE_F 0.11785113019775792f
#define SCALE_LOG2E (0.11785113019775792f * 1.4426950408889634f)

typedef __attribute__((ext_vector_type(16))) __bf16 v16bf;
typedef __attribute__((ext_vector_type(8)))  float  v8f;

union FragU { uint4 u[2]; v16bf v; };

__device__ __forceinline__ unsigned short f2bf(float f) {
  unsigned int u = __builtin_bit_cast(unsigned int, f);
  u += 0x7FFFu + ((u >> 16) & 1u);
  return (unsigned short)(u >> 16);
}

__device__ __forceinline__ v8f wmma_bf16(v16bf a, v16bf b, v8f c) {
  return __builtin_amdgcn_wmma_f32_16x16x32_bf16(false, a, false, b, (short)0, c, false, false);
}

// A-matrix 16x32 bf16 frag: lane holds row m=lane&15; lanes>=16 start K at +16.
// Two 16B loads: halves [kA..kA+7] and [kA+8..kA+15].
template<bool MASK>
__device__ __forceinline__ v16bf load_frag_a(const unsigned short* rowptr, int k0, int kmax, int lane) {
  FragU f;
  int kA = k0 + ((lane >> 4) << 4);
  int kB = kA + 8;
  f.u[0] = (!MASK || (kA + 8 <= kmax)) ? *(const uint4*)(rowptr + kA) : make_uint4(0u,0u,0u,0u);
  f.u[1] = (!MASK || (kB + 8 <= kmax)) ? *(const uint4*)(rowptr + kB) : make_uint4(0u,0u,0u,0u);
  return f.v;
}

// B-matrix 32x16 bf16 frag (NT: source row-major N x K, rowptr = row n=lane&15):
// lanes 0-15 hold K=k0..k0+15, lanes 16-31 hold K=k0+16..k0+31 (16 contiguous halves).
template<bool MASK>
__device__ __forceinline__ v16bf load_frag_b(const unsigned short* rowptr, int k0, int kmax, int lane) {
  FragU f;
  int kb = k0 + ((lane >> 4) << 4);
  f.u[0] = (!MASK || (kb + 8  <= kmax)) ? *(const uint4*)(rowptr + kb)     : make_uint4(0u,0u,0u,0u);
  f.u[1] = (!MASK || (kb + 16 <= kmax)) ? *(const uint4*)(rowptr + kb + 8) : make_uint4(0u,0u,0u,0u);
  return f.v;
}

// ---------------- fp32 -> bf16 conversion ----------------
__global__ void k_f32_to_bf16(const float* __restrict__ in, unsigned short* __restrict__ out, long long n) {
  long long i = (long long)blockIdx.x * blockDim.x + threadIdx.x;
  long long stride = (long long)gridDim.x * blockDim.x;
  for (; i < n; i += stride) out[i] = f2bf(in[i]);
}

// ---------------- generic NT GEMM: C[M,N] = A[M,K] * W[N,K]^T (+bias) ----------------
// One wave computes a 64x128 tile (4x8 WMMA frags). grid.x = tilesM*tilesN, grid.y = batch.
// N must be a multiple of 128; M tail rows are clamped on load, masked on store.
template<bool OUT_F32>
__global__ __launch_bounds__(32) void k_gemm_nt(
    const unsigned short* __restrict__ A, long long aBatch, int lda,
    const unsigned short* __restrict__ W, int ldw,
    const float* __restrict__ bias,
    void* __restrict__ Cv, long long cBatch, int ldc,
    int M, int N, int K, int tilesN)
{
  int lane = threadIdx.x & 31;
  int tile = blockIdx.x;
  int tm = tile / tilesN, tn = tile % tilesN;
  int b  = blockIdx.y;
  int m0 = tm * 64, n0 = tn * 128;
  const unsigned short* Ab = A + (long long)b * aBatch;

  const unsigned short* arow[4];
  const unsigned short* brow[8];
#pragma unroll
  for (int i = 0; i < 4; ++i) {
    int r = m0 + i * 16 + (lane & 15);
    r = min(r, M - 1);
    arow[i] = Ab + (long long)r * lda;
  }
#pragma unroll
  for (int j = 0; j < 8; ++j) {
    int n = n0 + j * 16 + (lane & 15);
    brow[j] = W + (long long)n * ldw;
  }

  v8f acc[4][8];
#pragma unroll
  for (int i = 0; i < 4; ++i)
#pragma unroll
    for (int j = 0; j < 8; ++j) acc[i][j] = v8f{0.f,0.f,0.f,0.f,0.f,0.f,0.f,0.f};

  for (int k0 = 0; k0 < K; k0 += 32) {
    v16bf af[4], bf[8];
#pragma unroll
    for (int i = 0; i < 4; ++i) af[i] = load_frag_a<false>(arow[i], k0, K, lane);
#pragma unroll
    for (int j = 0; j < 8; ++j) bf[j] = load_frag_b<false>(brow[j], k0, K, lane);
#pragma unroll
    for (int i = 0; i < 4; ++i)
#pragma unroll
      for (int j = 0; j < 8; ++j) acc[i][j] = wmma_bf16(af[i], bf[j], acc[i][j]);
  }

  int ncol = lane & 15;
  int mh   = (lane >> 4) << 3;  // lanes>=16 hold rows m+8
#pragma unroll
  for (int i = 0; i < 4; ++i) {
#pragma unroll
    for (int j = 0; j < 8; ++j) {
      v8f t = acc[i][j];
      int n = n0 + j * 16 + ncol;
#pragma unroll
      for (int e = 0; e < 8; ++e) {
        int m = m0 + i * 16 + e + mh;
        if (m < M) {
          float val = t[e];
          if (OUT_F32) {
            float* Cp = (float*)Cv + (long long)b * cBatch;
            if (bias) val += bias[n];
            Cp[(long long)m * ldc + n] = val;
          } else {
            unsigned short* Cp = (unsigned short*)Cv + (long long)b * cBatch;
            Cp[(long long)m * ldc + n] = f2bf(val);
          }
        }
      }
    }
  }
}

// ---------------- flash attention (bf16 WMMA, exp2-based fp32 softmax) ----------------
// grid.x = B * combosPerBatch (combo -> b, g, h); grid.y = q-slice of 64 rows.
// Block = 128 threads (4 waves); each wave owns 16 query rows.
__global__ __launch_bounds__(128) void k_flash_attn(
    const unsigned short* __restrict__ Qb, const unsigned short* __restrict__ Kb,
    const unsigned short* __restrict__ Vb, unsigned short* __restrict__ Ob,
    int combosPerBatch,
    long long qBatch, long long qGroup, int ldq,
    long long kBatch, long long kGroup, int ldk,
    long long vBatch, long long vGroup, int ldv,
    long long oBatch, long long oGroup, int ldo,
    int nQ, int nKV)
{
  __shared__ unsigned short sVT[80][128];       // V^T chunk, d rows (>=72 zero-padded)
  __shared__ unsigned short sP[4][16][128];     // per-wave P chunk

  int tid  = threadIdx.x;
  int lane = tid & 31;
  int w    = tid >> 5;

  int combo = blockIdx.x;
  int b = combo / combosPerBatch;
  int r = combo % combosPerBatch;
  int g = r >> 4;
  int h = r & 15;

  const unsigned short* Q = Qb + (long long)b * qBatch + (long long)g * qGroup + h * HD;
  const unsigned short* K = Kb + (long long)b * kBatch + (long long)g * kGroup + h * HD;
  const unsigned short* V = Vb + (long long)b * vBatch + (long long)g * vGroup + h * HD;
  unsigned short*       O = Ob + (long long)b * oBatch + (long long)g * oGroup + h * HD;

  int q0 = blockIdx.y * 64 + w * 16;
  int qr = min(q0 + (lane & 15), nQ - 1);
  const unsigned short* qrow = Q + (long long)qr * ldq;

  // Q fragments are chunk-invariant: hoist (3 K-steps covering Dh=72, zero padded).
  v16bf aq[3];
#pragma unroll
  for (int kidx = 0; kidx < 3; ++kidx) aq[kidx] = load_frag_a<true>(qrow, kidx * 32, HD, lane);

  v8f o[5];
#pragma unroll
  for (int jd = 0; jd < 5; ++jd) o[jd] = v8f{0.f,0.f,0.f,0.f,0.f,0.f,0.f,0.f};
  float rmax[8], rsum[8];
#pragma unroll
  for (int e = 0; e < 8; ++e) { rmax[e] = -1e30f; rsum[e] = 0.f; }

  int nChunks = (nKV + 127) >> 7;
  for (int c = 0; c < nChunks; ++c) {
    int nbase = c * 128;

    // ---- S = Q K^T (16 x 128), in units of SCALE*log2(e) ----
    v8f s[8];
#pragma unroll
    for (int j = 0; j < 8; ++j) s[j] = v8f{0.f,0.f,0.f,0.f,0.f,0.f,0.f,0.f};
#pragma unroll
    for (int kidx = 0; kidx < 3; ++kidx) {
      int k0 = kidx * 32;
#pragma unroll
      for (int j = 0; j < 8; ++j) {
        int kn = min(nbase + j * 16 + (lane & 15), nKV - 1);
        v16bf bfr = load_frag_b<true>(K + (long long)kn * ldk, k0, HD, lane);
        s[j] = wmma_bf16(aq[kidx], bfr, s[j]);
      }
    }

    // ---- scale to log2 domain + mask + chunk row-max ----
    float cmax[8];
#pragma unroll
    for (int e = 0; e < 8; ++e) cmax[e] = -1e30f;
#pragma unroll
    for (int j = 0; j < 8; ++j) {
      int n = nbase + j * 16 + (lane & 15);
      bool valid = (n < nKV);
#pragma unroll
      for (int e = 0; e < 8; ++e) {
        float v = s[j][e] * SCALE_LOG2E;
        if (!valid) v = -1e30f;
        s[j][e] = v;
        cmax[e] = fmaxf(cmax[e], v);
      }
    }
#pragma unroll
    for (int e = 0; e < 8; ++e) {
      cmax[e] = fmaxf(cmax[e], __shfl_xor(cmax[e], 1, 32));
      cmax[e] = fmaxf(cmax[e], __shfl_xor(cmax[e], 2, 32));
      cmax[e] = fmaxf(cmax[e], __shfl_xor(cmax[e], 4, 32));
      cmax[e] = fmaxf(cmax[e], __shfl_xor(cmax[e], 8, 32));
    }

    // ---- online softmax merge (exp2 domain) ----
    float alpha[8];
#pragma unroll
    for (int e = 0; e < 8; ++e) {
      float nm = fmaxf(rmax[e], cmax[e]);
      alpha[e] = exp2f(rmax[e] - nm);
      rmax[e] = nm;
    }
#pragma unroll
    for (int jd = 0; jd < 5; ++jd)
#pragma unroll
      for (int e = 0; e < 8; ++e) o[jd][e] *= alpha[e];

    float csum[8];
#pragma unroll
    for (int e = 0; e < 8; ++e) csum[e] = 0.f;
#pragma unroll
    for (int j = 0; j < 8; ++j)
#pragma unroll
      for (int e = 0; e < 8; ++e) {
        float p = exp2f(s[j][e] - rmax[e]);
        s[j][e] = p;
        csum[e] += p;
      }
#pragma unroll
    for (int e = 0; e < 8; ++e) {
      csum[e] += __shfl_xor(csum[e], 1, 32);
      csum[e] += __shfl_xor(csum[e], 2, 32);
      csum[e] += __shfl_xor(csum[e], 4, 32);
      csum[e] += __shfl_xor(csum[e], 8, 32);
      rsum[e] = rsum[e] * alpha[e] + csum[e];
    }

    // ---- write P chunk (bf16) to per-wave LDS ----
    {
      int mh = (lane >> 4) << 3;
#pragma unroll
      for (int j = 0; j < 8; ++j) {
        int n = j * 16 + (lane & 15);
#pragma unroll
        for (int e = 0; e < 8; ++e) sP[w][e + mh][n] = f2bf(s[j][e]);
      }
    }

    // ---- stage V^T chunk cooperatively ----
    for (int idx = tid; idx < 80 * 128; idx += 128) {
      int d  = idx >> 7;
      int nn = idx & 127;
      int vr = min(nbase + nn, nKV - 1);
      sVT[d][nn] = (d < HD) ? V[(long long)vr * ldv + d] : (unsigned short)0;
    }
    __syncthreads();

    // ---- O += P * V ----
#pragma unroll
    for (int kk = 0; kk < 128; kk += 32) {
      v16bf pa = load_frag_a<false>(&sP[w][lane & 15][0], kk, 128, lane);
#pragma unroll
      for (int jd = 0; jd < 5; ++jd) {
        v16bf vbf = load_frag_b<false>(&sVT[jd * 16 + (lane & 15)][0], kk, 128, lane);
        o[jd] = wmma_bf16(pa, vbf, o[jd]);
      }
    }
    __syncthreads();
  }

  // ---- normalize + store ----
  float inv[8];
#pragma unroll
  for (int e = 0; e < 8; ++e) inv[e] = 1.0f / rsum[e];
  int mh = (lane >> 4) << 3;
#pragma unroll
  for (int jd = 0; jd < 5; ++jd) {
    int d = jd * 16 + (lane & 15);
#pragma unroll
    for (int e = 0; e < 8; ++e) {
      int m = q0 + e + mh;
      if (m < nQ && d < HD)
        O[(long long)m * ldo + d] = f2bf(o[jd][e] * inv[e]);
    }
  }
}

// ---------------- host side ----------------
extern "C" void kernel_launch(void* const* d_in, const int* in_sizes, int n_in,
                              void* d_out, int out_size, void* d_ws, size_t ws_size,
                              hipStream_t stream) {
  const float* x      = (const float*)d_in[0];
  const float* w_qkv  = (const float*)d_in[1];
  const float* w_qa   = (const float*)d_in[2];
  const float* w_kv   = (const float*)d_in[3];
  const float* w_pn   = (const float*)d_in[4];
  const float* b_pn   = (const float*)d_in[5];
  const float* w_pa   = (const float*)d_in[6];
  const float* b_pa   = (const float*)d_in[7];
  float* out = (float*)d_out;

  // workspace carve (all bf16 = ushort), 256B aligned
  char* ws = (char*)d_ws;
  auto carve = [&](size_t elems) {
    unsigned short* p = (unsigned short*)ws;
    ws += ((elems * 2) + 255) & ~(size_t)255;
    return p;
  };
  const size_t S_X    = (size_t)BATCH * NTOK * C_DIM;
  const size_t S_WQKV = (size_t)3 * C_DIM * C_DIM;
  const size_t S_WSQ  = (size_t)C_DIM * C_DIM;
  const size_t S_WKV  = (size_t)2 * C_DIM * C_DIM;
  const size_t S_QKV  = (size_t)BATCH * NNON * 3 * C_DIM;
  const size_t S_QACT = (size_t)BATCH * TOK_ACT * C_DIM;
  const size_t S_KV   = (size_t)BATCH * NTOK * 2 * C_DIM;
  const size_t S_ANON = (size_t)BATCH * NNON * C_DIM;

  unsigned short* xbf   = carve(S_X);
  unsigned short* wqkvb = carve(S_WQKV);
  unsigned short* wqab  = carve(S_WSQ);
  unsigned short* wkvb  = carve(S_WKV);
  unsigned short* wpnb  = carve(S_WSQ);
  unsigned short* wpab  = carve(S_WSQ);
  unsigned short* qkvB  = carve(S_QKV);
  unsigned short* qactB = carve(S_QACT);
  unsigned short* kvB   = carve(S_KV);
  unsigned short* anonB = carve(S_ANON);
  unsigned short* aactB = carve(S_QACT);

  // 1) fp32 -> bf16
  k_f32_to_bf16<<<4096, 256, 0, stream>>>(x,     xbf,   (long long)S_X);
  k_f32_to_bf16<<<2048, 256, 0, stream>>>(w_qkv, wqkvb, (long long)S_WQKV);
  k_f32_to_bf16<<<1024, 256, 0, stream>>>(w_qa,  wqab,  (long long)S_WSQ);
  k_f32_to_bf16<<<1024, 256, 0, stream>>>(w_kv,  wkvb,  (long long)S_WKV);
  k_f32_to_bf16<<<1024, 256, 0, stream>>>(w_pn,  wpnb,  (long long)S_WSQ);
  k_f32_to_bf16<<<1024, 256, 0, stream>>>(w_pa,  wpab,  (long long)S_WSQ);

  // 2) input projections
  // qkv_non: A = x[:,57:] (768 x 1152) per batch, W = (3456 x 1152) -> (768 x 3456) bf16
  {
    int tM = NNON / 64, tN = (3 * C_DIM) / 128;
    k_gemm_nt<false><<<dim3(tM * tN, BATCH), 32, 0, stream>>>(
        xbf + (size_t)TOK_ACT * C_DIM, (long long)NTOK * C_DIM, C_DIM,
        wqkvb, C_DIM, nullptr,
        qkvB, (long long)NNON * 3 * C_DIM, 3 * C_DIM,
        NNON, 3 * C_DIM, C_DIM, tN);
  }
  // q_act: A = x[:,:57] (57 x 1152), W = (1152 x 1152)
  {
    int tM = 1, tN = C_DIM / 128;
    k_gemm_nt<false><<<dim3(tM * tN, BATCH), 32, 0, stream>>>(
        xbf, (long long)NTOK * C_DIM, C_DIM,
        wqab, C_DIM, nullptr,
        qactB, (long long)TOK_ACT * C_DIM, C_DIM,
        TOK_ACT, C_DIM, C_DIM, tN);
  }
  // kv_act: A = x (825 x 1152), W = (2304 x 1152)
  {
    int tM = (NTOK + 63) / 64, tN = (2 * C_DIM) / 128;
    k_gemm_nt<false><<<dim3(tM * tN, BATCH), 32, 0, stream>>>(
        xbf, (long long)NTOK * C_DIM, C_DIM,
        wkvb, C_DIM, nullptr,
        kvB, (long long)NTOK * 2 * C_DIM, 2 * C_DIM,
        NTOK, 2 * C_DIM, C_DIM, tN);
  }

  // 3) attention
  // non-act grouped: combos = 3*16 per batch, nQ=nKV=256, qkv packed layout (768 x 3456)
  k_flash_attn<<<dim3(BATCH * 48, 4), 128, 0, stream>>>(
      qkvB, qkvB + C_DIM, qkvB + 2 * C_DIM, anonB,
      48,
      (long long)NNON * 3 * C_DIM, (long long)GSZ * 3 * C_DIM, 3 * C_DIM,
      (long long)NNON * 3 * C_DIM, (long long)GSZ * 3 * C_DIM, 3 * C_DIM,
      (long long)NNON * 3 * C_DIM, (long long)GSZ * 3 * C_DIM, 3 * C_DIM,
      (long long)NNON * C_DIM, (long long)GSZ * C_DIM, C_DIM,
      GSZ, GSZ);
  // act: combos = 16 per batch, nQ=57, nKV=769 (tokens 56..824 of kv buffer)
  k_flash_attn<<<dim3(BATCH * 16, 1), 128, 0, stream>>>(
      qactB, kvB + (size_t)56 * 2 * C_DIM, kvB + (size_t)56 * 2 * C_DIM + C_DIM, aactB,
      16,
      (long long)TOK_ACT * C_DIM, 0, C_DIM,
      (long long)NTOK * 2 * C_DIM, 0, 2 * C_DIM,
      (long long)NTOK * 2 * C_DIM, 0, 2 * C_DIM,
      (long long)TOK_ACT * C_DIM, 0, C_DIM,
      TOK_ACT, NTOK - 56);

  // 4) output projections (fp32 out + bias), concat layout [act(57) | non(768)]
  {
    int tM = NNON / 64, tN = C_DIM / 128;
    k_gemm_nt<true><<<dim3(tM * tN, BATCH), 32, 0, stream>>>(
        anonB, (long long)NNON * C_DIM, C_DIM,
        wpnb, C_DIM, b_pn,
        (void*)(out + (size_t)TOK_ACT * C_DIM), (long long)NTOK * C_DIM, C_DIM,
        NNON, C_DIM, C_DIM, tN);
  }
  {
    int tM = 1, tN = C_DIM / 128;
    k_gemm_nt<true><<<dim3(tM * tN, BATCH), 32, 0, stream>>>(
        aactB, (long long)TOK_ACT * C_DIM, C_DIM,
        wpab, C_DIM, b_pa,
        (void*)out, (long long)NTOK * C_DIM, C_DIM,
        TOK_ACT, C_DIM, C_DIM, tN);
  }
}
